// MOELayer_16501264351755
// MI455X (gfx1250) — matrix operand order repaired and checked by reading, not compiled, not run on hardware
//
#include <hip/hip_runtime.h>
#include <hip/hip_bf16.h>
#include <math.h>

// ---------------------------------------------------------------------------
// MoE (E=8, top-2) SwiGLU FFN for MI455X / gfx1250.
// Grouped-GEMM: only selected (token, expert) pairs computed (103 GFLOP).
// bf16 WMMA (v_wmma_f32_16x16x32_bf16) with f32 accumulation.
// One-shot fp32->bf16 conversion pass puts all GEMM operands in bf16
// (weights 100MB -> L2-resident); tile fills are global_load_async_to_lds_b128
// into double-buffered LDS, overlapped with WMMA via ASYNCcnt.
// ---------------------------------------------------------------------------

typedef __attribute__((ext_vector_type(16))) __bf16 v16bf;
typedef __attribute__((ext_vector_type(8)))  float  v8f;

#define T_TOK   4096
#define H_DIM   1024
#define DFF_DIM 2048
#define N_EXP   8

// Workspace layout (byte offsets), total ~176 MB
#define WS_COUNTS 0ull                                   // 8 * int
#define WS_BASE   256ull                                 // 8 * int
#define WS_WT     512ull                                 // T*2 floats (32 KB)
#define WS_LIST   (512ull + 32ull*1024ull)               // E*T ints  (128 KB)
#define WS_HBUF   (512ull + 32ull*1024ull + 128ull*1024ull)  // 8192*2048 bf16
#define WS_YBUF   (WS_HBUF + 8192ull*2048ull*2ull)       // 8192*1024 f32
#define WS_XBF    (WS_YBUF + 8192ull*1024ull*4ull)       // T*H bf16 (8 MB)
#define WS_GWB    (WS_XBF + (size_t)T_TOK*H_DIM*2ull)    // E*DFF*H bf16 (33.5MB)
#define WS_UWB    (WS_GWB + 8ull*2048ull*1024ull*2ull)
#define WS_DWB    (WS_UWB + 8ull*2048ull*1024ull*2ull)

// ---------------------------------------------------------------------------
// helpers
// ---------------------------------------------------------------------------
__device__ __forceinline__ uint2 pack4_bf16(float4 v) {
  union { __bf16 h[4]; uint2 u; } p;
  p.h[0] = (__bf16)v.x; p.h[1] = (__bf16)v.y;
  p.h[2] = (__bf16)v.z; p.h[3] = (__bf16)v.w;
  return p.u;
}

// Generic pointer to shared memory: low 32 bits are the LDS byte offset
// (ISA 10.2).
__device__ __forceinline__ uint32_t lds_off(const void* p) {
  return (uint32_t)(uintptr_t)p;
}

// GVS-mode async copy: LDS[lds] = MEM[sbase + voff] (16 bytes). ASYNCcnt++.
__device__ __forceinline__ void async_copy_b128(uint32_t lds, uint32_t voff,
                                                uint64_t sbase) {
  asm volatile("global_load_async_to_lds_b128 %0, %1, %2"
               :: "v"(lds), "v"(voff), "s"(sbase) : "memory");
}
__device__ __forceinline__ void wait_async_le3() {
  asm volatile("s_wait_asynccnt 0x3" ::: "memory");
}
__device__ __forceinline__ void wait_async_le2() {
  asm volatile("s_wait_asynccnt 0x2" ::: "memory");
}
__device__ __forceinline__ void wait_async_le0() {
  asm volatile("s_wait_asynccnt 0x0" ::: "memory");
}

// A-operand fragment of a 16x32 bf16 tile (row-major in LDS, row stride 32).
// ISA 7.12.2: lanes 0-15 row M=lane hold K{0..7,16..23}; lanes 16-31 hold
// K{8..15,24..31}. Two 16B LDS reads per lane.
__device__ __forceinline__ v16bf load_a_frag(const __bf16* tile, int row, int half) {
  union { int4 i[2]; v16bf v; } u;
  const __bf16* p = tile + row * 32;
  u.i[0] = *(const int4*)(p + half * 8);
  u.i[1] = *(const int4*)(p + half * 8 + 16);
  return u.v;
}

// B-operand fragment: lane = column N; lanes 0-15 hold K{0..7,16..23}, lanes
// 16-31 hold K{8..15,24..31} (two 16B reads per lane).
__device__ __forceinline__ v16bf load_b_frag(const __bf16* tile, int row, int half) {
  union { int4 i[2]; v16bf v; } u;
  const __bf16* p = tile + row * 32;
  u.i[0] = *(const int4*)(p + half * 8);
  u.i[1] = *(const int4*)(p + half * 8 + 16);
  return u.v;
}

// ---------------------------------------------------------------------------
// 0) reset per-expert counters
// ---------------------------------------------------------------------------
__global__ void moe_zero_kernel(int* __restrict__ counts) {
  if (threadIdx.x < N_EXP) counts[threadIdx.x] = 0;
}

// ---------------------------------------------------------------------------
// 0b) fp32 -> bf16 bulk conversion (x and the three weight tensors)
// ---------------------------------------------------------------------------
__global__ __launch_bounds__(256) void moe_cvt_kernel(
    const float* __restrict__ src, __bf16* __restrict__ dst) {
  const size_t i = (size_t)blockIdx.x * 256 + threadIdx.x;   // float4 groups
  ((uint2*)dst)[i] = pack4_bf16(((const float4*)src)[i]);
}

// ---------------------------------------------------------------------------
// 1) router: sigmoid(x @ centroids^T) + bias, top-2, softmax; build per-expert
//    token lists. One wave per token (wave32), centroids staged in LDS.
// ---------------------------------------------------------------------------
__global__ __launch_bounds__(256) void moe_router_kernel(
    const float* __restrict__ x, const float* __restrict__ cent,
    const float* __restrict__ rbias, float* __restrict__ wt,
    int* __restrict__ list, int* __restrict__ counts)
{
  __shared__ float cs[N_EXP][H_DIM];                 // 32 KB
  for (int i = threadIdx.x; i < N_EXP * H_DIM / 4; i += 256)
    ((float4*)&cs[0][0])[i] = ((const float4*)cent)[i];
  __syncthreads();

  const int wave = threadIdx.x >> 5;
  const int lane = threadIdx.x & 31;
  const int tok  = blockIdx.x * 8 + wave;            // grid sized exactly

  const float* xr = x + (size_t)tok * H_DIM;
  float acc[N_EXP];
#pragma unroll
  for (int e = 0; e < N_EXP; ++e) acc[e] = 0.f;
  for (int h = lane; h < H_DIM; h += 32) {
    const float xv = xr[h];
#pragma unroll
    for (int e = 0; e < N_EXP; ++e) acc[e] += xv * cs[e][h];
  }
#pragma unroll
  for (int e = 0; e < N_EXP; ++e) {
#pragma unroll
    for (int m = 16; m >= 1; m >>= 1) acc[e] += __shfl_xor(acc[e], m, 32);
  }
  if (lane == 0) {
    float sc[N_EXP];
#pragma unroll
    for (int e = 0; e < N_EXP; ++e)
      sc[e] = 1.f / (1.f + __expf(-acc[e])) + rbias[e];
    int i0 = 0; float s0 = sc[0];
#pragma unroll
    for (int e = 1; e < N_EXP; ++e) if (sc[e] > s0) { s0 = sc[e]; i0 = e; }
    int i1 = (i0 == 0) ? 1 : 0; float s1 = sc[i1];
#pragma unroll
    for (int e = 0; e < N_EXP; ++e)
      if (e != i0 && sc[e] > s1) { s1 = sc[e]; i1 = e; }
    const float mx = fmaxf(s0, s1);
    const float e0 = __expf(s0 - mx), e1 = __expf(s1 - mx);
    const float inv = 1.f / (e0 + e1);
    wt[tok * 2 + 0] = e0 * inv;
    wt[tok * 2 + 1] = e1 * inv;
    const int slot0 = atomicAdd(&counts[i0], 1);
    list[(size_t)i0 * T_TOK + slot0] = tok * 2 + 0;
    const int slot1 = atomicAdd(&counts[i1], 1);
    list[(size_t)i1 * T_TOK + slot1] = tok * 2 + 1;
  }
}

// ---------------------------------------------------------------------------
// 2) exclusive prefix over counts + export counts to tail of d_out
// ---------------------------------------------------------------------------
__global__ void moe_prefix_kernel(const int* __restrict__ counts,
                                  int* __restrict__ base,
                                  float* __restrict__ out_counts) {
  if (threadIdx.x == 0) {
    int s = 0;
    for (int e = 0; e < N_EXP; ++e) { base[e] = s; s += counts[e]; }
  }
  if (threadIdx.x < N_EXP) out_counts[threadIdx.x] = (float)counts[threadIdx.x];
}

// ---------------------------------------------------------------------------
// 3) fused gate+up grouped GEMM + SwiGLU.  Block tile 64(M) x 64(N), K-step 32.
//    Double-buffered LDS filled by global_load_async_to_lds_b128 (3 x 16B per
//    thread per K step), overlapped with 4 WMMAs per wave per K step.
// ---------------------------------------------------------------------------
__global__ __launch_bounds__(256) void moe_gateup_kernel(
    const __bf16* __restrict__ xbf, const __bf16* __restrict__ gwb,
    const __bf16* __restrict__ uwb, const int* __restrict__ list,
    const int* __restrict__ counts, const int* __restrict__ base,
    __bf16* __restrict__ hbuf)
{
  const int e  = blockIdx.z;
  const int ne = counts[e];
  const int m0 = blockIdx.x * 64;
  if (m0 >= ne) return;                       // uniform early exit
  const int n0 = blockIdx.y * 64;
  const int be = base[e];
  const int* tl = list + (size_t)e * T_TOK;

  __shared__ alignas(16) __bf16 sX[2][64][32];
  __shared__ alignas(16) __bf16 sG[2][64][32];
  __shared__ alignas(16) __bf16 sU[2][64][32];
  __shared__ alignas(16) __bf16 sH[64][64];

  const int tid  = threadIdx.x;
  const int wid  = tid >> 5;
  const int lane = tid & 31;
  const int wm = wid & 3, wn = wid >> 2;
  const int rr = lane & 15, hf = lane >> 4;

  // --- K-invariant fill setup: 256 threads <-> 256 16B chunks per tile ---
  const int fr = tid >> 2;                    // row 0..63
  const int fc = tid & 3;                     // 16B chunk 0..3
  const int slot = min(m0 + fr, ne - 1);      // clamp: pad rows never stored
  const int tok  = tl[slot] >> 1;
  uint32_t vx = (uint32_t)((tok * H_DIM + fc * 8) * 2);         // byte offsets
  uint32_t vw = (uint32_t)(((n0 + fr) * H_DIM + fc * 8) * 2);
  const uint64_t sbx = (uint64_t)(uintptr_t)xbf;
  const uint64_t sbg = (uint64_t)(uintptr_t)(gwb + (size_t)e * DFF_DIM * H_DIM);
  const uint64_t sbu = (uint64_t)(uintptr_t)(uwb + (size_t)e * DFF_DIM * H_DIM);
  uint32_t lX[2], lG[2], lU[2];
#pragma unroll
  for (int b = 0; b < 2; ++b) {
    lX[b] = lds_off(&sX[b][fr][fc * 8]);
    lG[b] = lds_off(&sG[b][fr][fc * 8]);
    lU[b] = lds_off(&sU[b][fr][fc * 8]);
  }

  v8f cg0 = {0,0,0,0,0,0,0,0}, cg1 = {0,0,0,0,0,0,0,0};
  v8f cu0 = {0,0,0,0,0,0,0,0}, cu1 = {0,0,0,0,0,0,0,0};

  // prologue: batch 0 -> buffer 0
  async_copy_b128(lX[0], vx, sbx);
  async_copy_b128(lG[0], vw, sbg);
  async_copy_b128(lU[0], vw, sbu);
  vx += 64; vw += 64;

  const int NK = H_DIM / 32;                  // 32 K-steps
  for (int i = 0; i < NK; ++i) {
    __syncthreads();                          // buffer (i+1)&1 free
    if (i + 1 < NK) {
      const int nb = (i + 1) & 1;
      async_copy_b128(lX[nb], vx, sbx);
      async_copy_b128(lG[nb], vw, sbg);
      async_copy_b128(lU[nb], vw, sbu);
      vx += 64; vw += 64;
      wait_async_le3();                       // batch i done (in-order)
    } else {
      wait_async_le0();
    }
    __syncthreads();                          // batch i visible to all waves
    const int b = i & 1;
    const v16bf a  = load_a_frag(&sX[b][0][0], wm * 16 + rr, hf);
    const v16bf g0 = load_b_frag(&sG[b][0][0], wn * 32 + rr,      hf);
    const v16bf g1 = load_b_frag(&sG[b][0][0], wn * 32 + 16 + rr, hf);
    const v16bf u0 = load_b_frag(&sU[b][0][0], wn * 32 + rr,      hf);
    const v16bf u1 = load_b_frag(&sU[b][0][0], wn * 32 + 16 + rr, hf);
    cg0 = __builtin_amdgcn_wmma_f32_16x16x32_bf16(false, a, false, g0, (short)0, cg0, false, false);
    cg1 = __builtin_amdgcn_wmma_f32_16x16x32_bf16(false, a, false, g1, (short)0, cg1, false, false);
    cu0 = __builtin_amdgcn_wmma_f32_16x16x32_bf16(false, a, false, u0, (short)0, cu0, false, false);
    cu1 = __builtin_amdgcn_wmma_f32_16x16x32_bf16(false, a, false, u1, (short)0, cu1, false, false);
  }
  __syncthreads();
  // SwiGLU epilogue: C layout lane<16 -> (M=r, N=lane); lane>=16 -> (M=r+8)
#pragma unroll
  for (int j = 0; j < 2; ++j) {
    const v8f g = j ? cg1 : cg0;
    const v8f u = j ? cu1 : cu0;
#pragma unroll
    for (int r = 0; r < 8; ++r) {
      const float gv = g[r];
      const float hv = (gv / (1.f + __expf(-gv))) * u[r];
      sH[wm * 16 + r + hf * 8][wn * 32 + j * 16 + rr] = (__bf16)hv;
    }
  }
  __syncthreads();
  for (int i = tid; i < 64 * 16; i += 256) {
    const int r = i >> 4, c = i & 15;
    const int s2 = m0 + r;
    if (s2 < ne)
      *(uint2*)(hbuf + (size_t)(be + s2) * DFF_DIM + n0 + c * 4) =
          *(const uint2*)&sH[r][c * 4];
  }
}

// ---------------------------------------------------------------------------
// 4) down-projection grouped GEMM: ybuf[tok*2+k][H] = h @ down_w[e]^T (f32)
// ---------------------------------------------------------------------------
__global__ __launch_bounds__(256) void moe_down_kernel(
    const __bf16* __restrict__ hbuf, const __bf16* __restrict__ dwb,
    const int* __restrict__ list, const int* __restrict__ counts,
    const int* __restrict__ base, float* __restrict__ ybuf)
{
  const int e  = blockIdx.z;
  const int ne = counts[e];
  const int m0 = blockIdx.x * 64;
  if (m0 >= ne) return;
  const int n0 = blockIdx.y * 64;
  const int be = base[e];
  const int* tl = list + (size_t)e * T_TOK;

  __shared__ alignas(16) __bf16 sA[2][64][32];
  __shared__ alignas(16) __bf16 sW[2][64][32];
  __shared__ alignas(16) float  sY[64][64];

  const int tid  = threadIdx.x;
  const int wid  = tid >> 5;
  const int lane = tid & 31;
  const int wm = wid & 3, wn = wid >> 2;
  const int rr = lane & 15, hf = lane >> 4;

  const int fr = tid >> 2;
  const int fc = tid & 3;
  const int slot = min(m0 + fr, ne - 1);
  uint32_t va = (uint32_t)(((be + slot) * DFF_DIM + fc * 8) * 2);
  uint32_t vw = (uint32_t)(((n0 + fr) * DFF_DIM + fc * 8) * 2);
  const uint64_t sba = (uint64_t)(uintptr_t)hbuf;
  const uint64_t sbw = (uint64_t)(uintptr_t)(dwb + (size_t)e * H_DIM * DFF_DIM);
  uint32_t lA[2], lW[2];
#pragma unroll
  for (int b = 0; b < 2; ++b) {
    lA[b] = lds_off(&sA[b][fr][fc * 8]);
    lW[b] = lds_off(&sW[b][fr][fc * 8]);
  }

  v8f c0 = {0,0,0,0,0,0,0,0}, c1 = {0,0,0,0,0,0,0,0};

  async_copy_b128(lA[0], va, sba);
  async_copy_b128(lW[0], vw, sbw);
  va += 64; vw += 64;

  const int NK = DFF_DIM / 32;                // 64 K-steps
  for (int i = 0; i < NK; ++i) {
    __syncthreads();
    if (i + 1 < NK) {
      const int nb = (i + 1) & 1;
      async_copy_b128(lA[nb], va, sba);
      async_copy_b128(lW[nb], vw, sbw);
      va += 64; vw += 64;
      wait_async_le2();
    } else {
      wait_async_le0();
    }
    __syncthreads();
    const int b = i & 1;
    const v16bf a  = load_a_frag(&sA[b][0][0], wm * 16 + rr, hf);
    const v16bf b0 = load_b_frag(&sW[b][0][0], wn * 32 + rr,      hf);
    const v16bf b1 = load_b_frag(&sW[b][0][0], wn * 32 + 16 + rr, hf);
    c0 = __builtin_amdgcn_wmma_f32_16x16x32_bf16(false, a, false, b0, (short)0, c0, false, false);
    c1 = __builtin_amdgcn_wmma_f32_16x16x32_bf16(false, a, false, b1, (short)0, c1, false, false);
  }
  __syncthreads();
#pragma unroll
  for (int j = 0; j < 2; ++j) {
    const v8f cc = j ? c1 : c0;
#pragma unroll
    for (int r = 0; r < 8; ++r)
      sY[wm * 16 + r + hf * 8][wn * 32 + j * 16 + rr] = cc[r];
  }
  __syncthreads();
  for (int i = tid; i < 64 * 16; i += 256) {
    const int r = i >> 4, c = i & 15;
    const int s2 = m0 + r;
    if (s2 < ne) {
      const int pr = tl[s2];                       // tok*2 + k
      *(float4*)(ybuf + (size_t)pr * H_DIM + n0 + c * 4) =
          *(const float4*)&sY[r][c * 4];
    }
  }
}

// ---------------------------------------------------------------------------
// 5) combine: out[t] = w0*y[t,0] + w1*y[t,1]
// ---------------------------------------------------------------------------
__global__ __launch_bounds__(256) void moe_combine_kernel(
    const float* __restrict__ ybuf, const float* __restrict__ wt,
    float* __restrict__ out)
{
  const size_t i = (size_t)blockIdx.x * 256 + threadIdx.x;   // over T*H/4
  const size_t t = i / (H_DIM / 4);
  const size_t c = i % (H_DIM / 4);
  const float w0 = wt[t * 2 + 0], w1 = wt[t * 2 + 1];
  const float4 a = ((const float4*)(ybuf + (t * 2 + 0) * H_DIM))[c];
  const float4 b = ((const float4*)(ybuf + (t * 2 + 1) * H_DIM))[c];
  float4 r;
  r.x = w0 * a.x + w1 * b.x;  r.y = w0 * a.y + w1 * b.y;
  r.z = w0 * a.z + w1 * b.z;  r.w = w0 * a.w + w1 * b.w;
  ((float4*)out)[i] = r;
}

// ---------------------------------------------------------------------------
extern "C" void kernel_launch(void* const* d_in, const int* in_sizes, int n_in,
                              void* d_out, int out_size, void* d_ws, size_t ws_size,
                              hipStream_t stream)
{
  const float* x     = (const float*)d_in[0];
  const float* cent  = (const float*)d_in[1];
  const float* gw    = (const float*)d_in[2];
  const float* uw    = (const float*)d_in[3];
  const float* dw    = (const float*)d_in[4];
  const float* rbias = (const float*)d_in[5];
  float* out = (float*)d_out;

  char* ws = (char*)d_ws;
  int*    counts = (int*)(ws + WS_COUNTS);
  int*    base   = (int*)(ws + WS_BASE);
  float*  wt     = (float*)(ws + WS_WT);
  int*    list   = (int*)(ws + WS_LIST);
  __bf16* hbuf   = (__bf16*)(ws + WS_HBUF);
  float*  ybuf   = (float*)(ws + WS_YBUF);
  __bf16* xbf    = (__bf16*)(ws + WS_XBF);
  __bf16* gwb    = (__bf16*)(ws + WS_GWB);
  __bf16* uwb    = (__bf16*)(ws + WS_UWB);
  __bf16* dwb    = (__bf16*)(ws + WS_DWB);

  moe_zero_kernel<<<1, 32, 0, stream>>>(counts);

  const int wblk = (N_EXP * DFF_DIM * H_DIM / 4) / 256;   // 16384
  moe_cvt_kernel<<<(T_TOK * H_DIM / 4) / 256, 256, 0, stream>>>(x, xbf);
  moe_cvt_kernel<<<wblk, 256, 0, stream>>>(gw, gwb);
  moe_cvt_kernel<<<wblk, 256, 0, stream>>>(uw, uwb);
  moe_cvt_kernel<<<wblk, 256, 0, stream>>>(dw, dwb);

  moe_router_kernel<<<T_TOK / 8, 256, 0, stream>>>(x, cent, rbias, wt, list, counts);
  moe_prefix_kernel<<<1, 32, 0, stream>>>(counts, base, out + (size_t)T_TOK * H_DIM);
  moe_gateup_kernel<<<dim3(T_TOK / 64, DFF_DIM / 64, N_EXP), 256, 0, stream>>>(
      xbf, gwb, uwb, list, counts, base, hbuf);
  moe_down_kernel<<<dim3(T_TOK / 64, H_DIM / 64, N_EXP), 256, 0, stream>>>(
      hbuf, dwb, list, counts, base, ybuf);
  moe_combine_kernel<<<(T_TOK * H_DIM / 4) / 256, 256, 0, stream>>>(ybuf, wt, out);
}
